// Wave2D_69793218560502
// MI455X (gfx1250) — compile-verified, matrix operand order checked by reading, not run on hardware
//
#include <hip/hip_runtime.h>

// ---------------------------------------------------------------------------
// Wave2D on MI455X (gfx1250, wave32, WMMA). All channel contractions and all
// 32-point DCT/IDCT transforms are mapped onto v_wmma_f32_16x16x32_bf16
// (bf16 inputs, f32 accumulate). GEMM K-tiles are double-buffered in LDS via
// global_load_async_to_lds_b128 (ASYNCcnt); the DCT kernel's 32x32 image
// loads use the Tensor Data Mover (TENSORcnt) with LDS padding done by the
// TDM pad feature. Elementwise stages stay in f32 VALU.
// ---------------------------------------------------------------------------

typedef __attribute__((ext_vector_type(16))) __bf16 v16bf;
typedef __attribute__((ext_vector_type(8)))  float  v8f;
typedef __attribute__((ext_vector_type(4)))  unsigned int u32x4;
typedef __attribute__((ext_vector_type(8)))  int          i32x8;
typedef __attribute__((ext_vector_type(4)))  int          i32x4;

// typed address-space pointers for the async-load builtin
typedef __attribute__((address_space(1))) i32x4* gptr_i4;
typedef __attribute__((address_space(3))) i32x4* lptr_i4;

#define PI_F 3.14159265358979323846f

#if __has_builtin(__builtin_amdgcn_global_load_async_to_lds_b128)
#define ASYNC_LDS 1
#else
#define ASYNC_LDS 0
#endif

#if __has_builtin(__builtin_amdgcn_tensor_load_to_lds)
#define HAS_TDM 1
#else
#define HAS_TDM 0
#endif

// 16B global -> LDS copy; async (ASYNCcnt) when the gfx1250 builtin exists.
__device__ inline void copy16_g2l(const void* g, void* l) {
#if ASYNC_LDS
  __builtin_amdgcn_global_load_async_to_lds_b128(
      (gptr_i4)(uintptr_t)g, (lptr_i4)l, 0, 0);
#else
  *reinterpret_cast<uint4*>(l) = *reinterpret_cast<const uint4*>(g);
#endif
}

template <int N>
__device__ inline void async_wait() {
#if ASYNC_LDS
#if __has_builtin(__builtin_amdgcn_s_wait_asynccnt)
  __builtin_amdgcn_s_wait_asynccnt((short)N);
#else
  asm volatile("s_wait_asynccnt %0" ::"n"(N) : "memory");
#endif
#endif
}

// --- fragment loaders --------------------------------------------------------
// ISA 16-bit 16x32 A layout: element e of lane L (half = L>>4) holds
// K = (e/8)*16 + half*8 + (e%8) of row (L&15). B (32x16) is symmetric with
// lanes holding columns; our B source is W(N,K) row-major so lane n reads
// row n of W contiguously.
__device__ inline v16bf frag_bf16(const __bf16* rowp, int half) {
  v16bf f;
#pragma unroll
  for (int e = 0; e < 8; ++e) {
    f[e]     = rowp[half * 8 + e];
    f[e + 8] = rowp[16 + half * 8 + e];
  }
  return f;
}

__device__ inline v16bf frag_f32(const float* rowp, int half) {
  v16bf f;
#pragma unroll
  for (int e = 0; e < 8; ++e) {
    f[e]     = (__bf16)rowp[half * 8 + e];
    f[e + 8] = (__bf16)rowp[16 + half * 8 + e];
  }
  return f;
}

// --- kernel: f32 -> bf16 conversion (weights / freq_embed) -------------------
__global__ void cvt_bf16_kernel(const float* __restrict__ in,
                                __bf16* __restrict__ out, int n) {
  int i = blockIdx.x * 256 + threadIdx.x;
  if (i < n) out[i] = (__bf16)in[i];
}

// --- kernel: depthwise 3x3 conv, NCHW f32 -> token-major bf16 ----------------
__global__ __launch_bounds__(256)
void dwconv_kernel(const float* __restrict__ x, const float* __restrict__ w,
                   const float* __restrict__ bias, __bf16* __restrict__ out) {
  int idx = blockIdx.x * 256 + threadIdx.x;        // b*2^19 + c*1024 + h*32 + w
  int wc = idx & 31;
  int h  = (idx >> 5) & 31;
  int c  = (idx >> 10) & 511;
  int b  = idx >> 19;
  const float* img = x + ((size_t)idx & ~(size_t)1023);
  float acc = bias[c];
#pragma unroll
  for (int kh = 0; kh < 3; ++kh) {
    int hh = h + kh - 1;
    if (hh < 0 || hh > 31) continue;
#pragma unroll
    for (int kw = 0; kw < 3; ++kw) {
      int ww = wc + kw - 1;
      if (ww < 0 || ww > 31) continue;
      acc += img[hh * 32 + ww] * w[c * 9 + kh * 3 + kw];
    }
  }
  out[((size_t)(b << 10) + h * 32 + wc) * 512 + c] = (__bf16)acc;
}

// --- kernel: WMMA GEMM, out[m,n] = sum_k A[m,k] * W[n,k] + bias[n] -----------
// 256 threads = 8 waves. WG tile 128x128. K stepped 32 at a time through
// double-buffered LDS tiles filled by async global->LDS copies; compute on
// tile k overlaps the async fill of tile k+1 (wait ASYNCcnt<=4 keeps the
// newest 4 copies in flight).
// MODE 0: lin    (N=1024): n<512 -> xs (bf16 token + f32 NCHW); n>=512 -> z f32
// MODE 1: vel    (N=512):  f32 NCHW
// MODE 2: token  (M=1024): relu -> cos/sin tables (C,HW)
// MODE 3: outproj(N=512):  f32 NCHW into d_out
template <int MODE>
__global__ __launch_bounds__(256)
void gemm_wmma(const __bf16* __restrict__ A, const __bf16* __restrict__ Wt,
               const float* __restrict__ bias,
               float* __restrict__ out0, float* __restrict__ out1,
               __bf16* __restrict__ outb, const float* __restrict__ cPtr) {
  constexpr int K = 512;
  __shared__ __align__(16) __bf16 As[2][128 * 32];
  __shared__ __align__(16) __bf16 Bs[2][128 * 32];

  const int tid  = threadIdx.x;
  const int lane = tid & 31;
  const int wid  = tid >> 5;
  const int wr   = wid & 3;
  const int wcol = wid >> 2;
  const int l16  = lane & 15;
  const int half = lane >> 4;
  const int mBase = blockIdx.x * 128;
  const int nBase = blockIdx.y * 128;

  v8f acc[2][4];
#pragma unroll
  for (int i = 0; i < 2; ++i)
#pragma unroll
    for (int j = 0; j < 4; ++j)
#pragma unroll
      for (int r = 0; r < 8; ++r) acc[i][j][r] = 0.0f;

  auto issueTile = [&](int kb, int buf) {
#pragma unroll
    for (int i = 0; i < 2; ++i) {
      int chunk = tid + i * 256;      // 512 chunks of 8 bf16 per tile
      int row = chunk >> 2;
      int c8  = chunk & 3;
      copy16_g2l(&A[(size_t)(mBase + row) * K + kb + c8 * 8],
                 &As[buf][row * 32 + c8 * 8]);
      copy16_g2l(&Wt[(size_t)(nBase + row) * K + kb + c8 * 8],
                 &Bs[buf][row * 32 + c8 * 8]);
    }
  };

  issueTile(0, 0);
  for (int it = 0; it < K / 32; ++it) {
    const int buf = it & 1;
    if (it + 1 < K / 32) {
      issueTile((it + 1) * 32, buf ^ 1);   // prefetch next tile (4 newer ops)
      async_wait<4>();                     // current tile's 4 copies complete
    } else {
      async_wait<0>();
    }
    __syncthreads();

    v16bf af[2], bfr[4];
#pragma unroll
    for (int i = 0; i < 2; ++i)
      af[i] = frag_bf16(&As[buf][(wr * 32 + i * 16 + l16) * 32], half);
#pragma unroll
    for (int j = 0; j < 4; ++j)
      bfr[j] = frag_bf16(&Bs[buf][(wcol * 64 + j * 16 + l16) * 32], half);
#pragma unroll
    for (int i = 0; i < 2; ++i)
#pragma unroll
      for (int j = 0; j < 4; ++j)
        acc[i][j] = __builtin_amdgcn_wmma_f32_16x16x32_bf16(
            false, af[i], false, bfr[j], (short)0, acc[i][j], false, false);
    __syncthreads();
  }

  float cv = 0.0f;
  if constexpr (MODE == 2) cv = cPtr[0];

#pragma unroll
  for (int i = 0; i < 2; ++i)
#pragma unroll
    for (int j = 0; j < 4; ++j)
#pragma unroll
      for (int r = 0; r < 8; ++r) {
        int m = mBase + wr * 32 + i * 16 + half * 8 + r;
        int n = nBase + wcol * 64 + j * 16 + l16;
        float v = acc[i][j][r] + bias[n];
        if constexpr (MODE == 0) {
          int b = m >> 10, hw = m & 1023;
          if (n < 512) {
            outb[(size_t)m * 512 + n] = (__bf16)v;              // xs bf16 tokens
            out0[((size_t)(b * 512 + n)) * 1024 + hw] = v;      // xs f32 NCHW
          } else {
            out1[(size_t)m * 512 + (n - 512)] = v;              // z f32 tokens
          }
        } else if constexpr (MODE == 1) {
          int b = m >> 10, hw = m & 1023;
          out0[((size_t)(b * 512 + n)) * 1024 + hw] = v;        // vel f32 NCHW
        } else if constexpr (MODE == 2) {
          float t = fmaxf(v, 0.0f);                             // relu
          float a = cv * t;
          out0[(size_t)n * 1024 + m] = cosf(a);                 // cosT (C,HW)
          out1[(size_t)n * 1024 + m] = sinf(a) / (cv + 1e-6f);  // sinT (C,HW)
        } else {
          int b = m >> 10, hw = m & 1023;
          out0[((size_t)(b * 512 + n)) * 1024 + hw] = v;        // final NCHW
        }
      }
}

// --- DCT helpers -------------------------------------------------------------
__device__ inline void mm32(const float* Ab, int As, const float* Bb, int Bs,
                            int l16, int half, v8f (&acc)[2][2]) {
  v16bf af[2], bfr[2];
#pragma unroll
  for (int i = 0; i < 2; ++i) af[i] = frag_f32(Ab + (i * 16 + l16) * As, half);
#pragma unroll
  for (int j = 0; j < 2; ++j) bfr[j] = frag_f32(Bb + (j * 16 + l16) * Bs, half);
#pragma unroll
  for (int i = 0; i < 2; ++i)
#pragma unroll
    for (int j = 0; j < 2; ++j)
      acc[i][j] = __builtin_amdgcn_wmma_f32_16x16x32_bf16(
          false, af[i], false, bfr[j], (short)0, acc[i][j], false, false);
}

__device__ inline void zero22(v8f (&a)[2][2]) {
#pragma unroll
  for (int i = 0; i < 2; ++i)
#pragma unroll
    for (int j = 0; j < 2; ++j)
#pragma unroll
      for (int r = 0; r < 8; ++r) a[i][j][r] = 0.0f;
}

__device__ inline void storeT36(float* T, const v8f (&acc)[2][2], int l16, int half) {
#pragma unroll
  for (int i = 0; i < 2; ++i)
#pragma unroll
    for (int j = 0; j < 2; ++j)
#pragma unroll
      for (int r = 0; r < 8; ++r) {
        int m = i * 16 + half * 8 + r;
        int n = j * 16 + l16;
        T[n * 36 + m] = acc[i][j][r];   // transposed store
      }
}

// Load one 32x32 f32 image (contiguous 4KB in global) into a padded
// 32x36-float LDS buffer. TDM path: one tensor_load_to_lds with a 2D D#
// (tile 32x32, data_size=4B) using pad_enable(interval=32 dw, amount=4 dw)
// to produce the 36-float row stride. Fallback: per-lane float4 copies.
__device__ inline void load_image(const float* src, float* X, int lane) {
#if HAS_TDM
  unsigned long long ga = (unsigned long long)(uintptr_t)src;
  unsigned int la =
      (unsigned int)(uintptr_t)(__attribute__((address_space(3))) void*)X;
  u32x4 g0;
  g0[0] = 1u;                                    // count=1 (valid user D#)
  g0[1] = la;                                    // lds_addr (bytes)
  g0[2] = (unsigned int)ga;                      // global_addr[31:0]
  g0[3] = (unsigned int)((ga >> 32) & 0x1FFFFFFu) | (2u << 30);  // [56:32]+type=2
  i32x8 g1;
  g1[0] = (2 << 16) | (1 << 20) | (4 << 22) | (3 << 25);
  //        data_size=4B | pad_en | pad every 32 DW | pad 4 DW
  g1[1] = (32 << 16);                            // tensor_dim0 = 32
  g1[2] = (32 << 16);                            // tensor_dim1 = 32
  g1[3] = (32 << 16);                            // tile_dim0 = 32
  g1[4] = 32;                                    // tile_dim1 = 32
  g1[5] = 32;                                    // tensor_dim0_stride = 32
  g1[6] = (1024 << 16);                          // tensor_dim1_stride = 1024
  g1[7] = 0;
  i32x4 z4 = {0, 0, 0, 0};
  i32x8 z8 = {0, 0, 0, 0, 0, 0, 0, 0};
  __builtin_amdgcn_tensor_load_to_lds(g0, g1, z4, z4, z8, 0);
  (void)lane;
#else
#pragma unroll
  for (int i = 0; i < 8; ++i) {
    int e = i * 128 + lane * 4;
    int row = e >> 5, col = e & 31;
    float4 v = *reinterpret_cast<const float4*>(src + e);
    *reinterpret_cast<float4*>(&X[row * 36 + col]) = v;
  }
#endif
}

__device__ inline void load_image_wait() {
#if HAS_TDM
#if __has_builtin(__builtin_amdgcn_s_wait_tensorcnt)
  __builtin_amdgcn_s_wait_tensorcnt((short)0);
#else
  asm volatile("s_wait_tensorcnt 0x0" ::: "memory");
#endif
#endif
}

// --- kernel: per-(b,c) image DCT2D(xs), DCT2D(vel), decay, spectral combine,
//     IDCT2D -> xf token-major. One wave per image; 4 images per workgroup. ---
__global__ __launch_bounds__(128)
void dct_combine_kernel(const float* __restrict__ xsN, const float* __restrict__ xvN,
                        const float* __restrict__ cosT, const float* __restrict__ sinT,
                        const float* __restrict__ alphaPtr, float* __restrict__ xf) {
  __shared__ __align__(16) float Dm[32 * 32];       // D[n][k] ortho DCT-II
  __shared__ __align__(16) float Dt[32 * 32];       // Dt[k][n] = D[n][k]
  __shared__ __align__(16) float Xb[4][32 * 36];    // padded 36-float stride
  __shared__ __align__(16) float Tb[4][32 * 36];

  const int tid = threadIdx.x, lane = tid & 31, wid = tid >> 5;
  const int l16 = lane & 15, half = lane >> 4;

  for (int idx = tid; idx < 1024; idx += 128) {
    int n = idx >> 5, k = idx & 31;
    float s = (n == 0) ? 0.17677669529663688f /*1/sqrt(32)*/ : 0.25f /*sqrt(2/32)*/;
    float v = s * cosf(PI_F * (float)((2 * k + 1) * n) / 64.0f);
    Dm[n * 32 + k] = v;
    Dt[k * 32 + n] = v;
  }
  __syncthreads();

  const int img = blockIdx.x * 4 + wid;   // == b*512 + c
  const int b   = img >> 9;
  const int cch = img & 511;
  float* X = Xb[wid];
  float* T = Tb[wid];
  const float halfAlpha = 0.5f * alphaPtr[0];

  v8f U[2][2], V[2][2], P[2][2];

  // ---- U = decay .* (D * Xs * D^T) ----
  load_image(xsN + (size_t)img * 1024, X, lane);
  load_image_wait();
  __syncthreads();
  zero22(P); mm32(X, 36, Dm, 32, l16, half, P);          // P = Xs * D^T
  storeT36(T, P, l16, half);
  __syncthreads();
  zero22(U); mm32(Dm, 32, T, 36, l16, half, U);          // U = D * P
#pragma unroll
  for (int i = 0; i < 2; ++i)
#pragma unroll
    for (int j = 0; j < 2; ++j)
#pragma unroll
      for (int r = 0; r < 8; ++r) {
        int M = i * 16 + half * 8 + r, N = j * 16 + l16;
        float wn = PI_F * (float)M * (1.0f / 32.0f);
        float wm = PI_F * (float)N * (1.0f / 32.0f);
        U[i][j][r] *= expf(-(wn * wn + wm * wm));
      }
  __syncthreads();

  // ---- V = decay .* (D * Xv * D^T) ----
  load_image(xvN + (size_t)img * 1024, X, lane);
  load_image_wait();
  __syncthreads();
  zero22(P); mm32(X, 36, Dm, 32, l16, half, P);
  storeT36(T, P, l16, half);
  __syncthreads();
  zero22(V); mm32(Dm, 32, T, 36, l16, half, V);
#pragma unroll
  for (int i = 0; i < 2; ++i)
#pragma unroll
    for (int j = 0; j < 2; ++j)
#pragma unroll
      for (int r = 0; r < 8; ++r) {
        int M = i * 16 + half * 8 + r, N = j * 16 + l16;
        float wn = PI_F * (float)M * (1.0f / 32.0f);
        float wm = PI_F * (float)N * (1.0f / 32.0f);
        V[i][j][r] *= expf(-(wn * wn + wm * wm));
      }
  __syncthreads();

  // ---- F = cosT.*U + sinT.*(V + 0.5*alpha*U)  (into X, row-major) ----
#pragma unroll
  for (int i = 0; i < 2; ++i)
#pragma unroll
    for (int j = 0; j < 2; ++j)
#pragma unroll
      for (int r = 0; r < 8; ++r) {
        int M = i * 16 + half * 8 + r, N = j * 16 + l16;
        size_t fi = (size_t)cch * 1024 + M * 32 + N;
        float u = U[i][j][r], v = V[i][j][r];
        X[M * 36 + N] = cosT[fi] * u + sinT[fi] * (v + halfAlpha * u);
      }
  __syncthreads();

  // ---- Y = D^T * F * D ----
  zero22(P); mm32(X, 36, Dt, 32, l16, half, P);          // Q = F * D
  storeT36(T, P, l16, half);
  __syncthreads();
  zero22(U); mm32(Dt, 32, T, 36, l16, half, U);          // Y = D^T * Q

#pragma unroll
  for (int i = 0; i < 2; ++i)
#pragma unroll
    for (int j = 0; j < 2; ++j)
#pragma unroll
      for (int r = 0; r < 8; ++r) {
        int M = i * 16 + half * 8 + r, N = j * 16 + l16;  // M=h, N=w
        xf[((size_t)(b << 10) + M * 32 + N) * 512 + cch] = U[i][j][r];
      }
}

// --- kernel: LayerNorm over C (wave per token) + SiLU gate -> bf16 -----------
__global__ __launch_bounds__(256)
void ln_gate_kernel(const float* __restrict__ xf, const float* __restrict__ z,
                    const float* __restrict__ gamma, const float* __restrict__ beta,
                    __bf16* __restrict__ gated) {
  const int lane = threadIdx.x & 31, wid = threadIdx.x >> 5;
  const int token = blockIdx.x * 8 + wid;
  const float* row = xf + (size_t)token * 512;
  const float* zr  = z  + (size_t)token * 512;

  float vals[16];
  float s = 0.0f, s2 = 0.0f;
#pragma unroll
  for (int i = 0; i < 16; ++i) {
    float v = row[i * 32 + lane];
    vals[i] = v;
    s += v;
    s2 += v * v;
  }
#pragma unroll
  for (int off = 16; off > 0; off >>= 1) {
    s  += __shfl_xor(s, off, 32);
    s2 += __shfl_xor(s2, off, 32);
  }
  float mu  = s * (1.0f / 512.0f);
  float var = s2 * (1.0f / 512.0f) - mu * mu;
  float inv = rsqrtf(var + 1e-5f);
#pragma unroll
  for (int i = 0; i < 16; ++i) {
    int c = i * 32 + lane;
    float ln = (vals[i] - mu) * inv * gamma[c] + beta[c];
    float zv = zr[c];
    float silu = zv / (1.0f + expf(-zv));
    gated[(size_t)token * 512 + c] = (__bf16)(ln * silu);
  }
}

// ---------------------------------------------------------------------------
extern "C" void kernel_launch(void* const* d_in, const int* in_sizes, int n_in,
                              void* d_out, int out_size, void* d_ws, size_t ws_size,
                              hipStream_t stream) {
  (void)in_sizes; (void)n_in; (void)out_size; (void)ws_size;
  const float* x      = (const float*)d_in[0];
  const float* freq   = (const float*)d_in[1];
  const float* dw_w   = (const float*)d_in[2];
  const float* dw_b   = (const float*)d_in[3];
  const float* lin_w  = (const float*)d_in[4];
  const float* lin_b  = (const float*)d_in[5];
  const float* vel_w  = (const float*)d_in[6];
  const float* vel_b  = (const float*)d_in[7];
  const float* tok_w  = (const float*)d_in[8];
  const float* tok_b  = (const float*)d_in[9];
  const float* gamma  = (const float*)d_in[10];
  const float* beta   = (const float*)d_in[11];
  const float* out_w  = (const float*)d_in[12];
  const float* out_b  = (const float*)d_in[13];
  const float* cptr   = (const float*)d_in[14];
  const float* aptr   = (const float*)d_in[15];

  char* ws = (char*)d_ws;
  size_t off = 0;
  auto take = [&](size_t bytes) {
    size_t r = off;
    off += (bytes + 255) & ~(size_t)255;
    return r;
  };
  const size_t M = 16384;      // B*H*W tokens
  __bf16* xc    = (__bf16*)(ws + take(M * 512 * 2));
  __bf16* xs_b  = (__bf16*)(ws + take(M * 512 * 2));
  float*  xs_n  = (float* )(ws + take(M * 512 * 4));
  float*  zbuf  = (float* )(ws + take(M * 512 * 4));
  float*  xv_n  = (float* )(ws + take(M * 512 * 4));
  float*  cosT  = (float* )(ws + take(512 * 1024 * 4));
  float*  sinT  = (float* )(ws + take(512 * 1024 * 4));
  float*  xf    = (float* )(ws + take(M * 512 * 4));
  __bf16* wlin  = (__bf16*)(ws + take(524288 * 2));
  __bf16* wvel  = (__bf16*)(ws + take(262144 * 2));
  __bf16* wtok  = (__bf16*)(ws + take(262144 * 2));
  __bf16* wout  = (__bf16*)(ws + take(262144 * 2));
  __bf16* freqb = (__bf16*)(ws + take(524288 * 2));
  __bf16* gated = xc;  // conv output dead after GEMM1; reuse for gated tokens

  // weights / embeddings -> bf16
  cvt_bf16_kernel<<<(524288 + 255) / 256, 256, 0, stream>>>(lin_w, wlin, 524288);
  cvt_bf16_kernel<<<(262144 + 255) / 256, 256, 0, stream>>>(vel_w, wvel, 262144);
  cvt_bf16_kernel<<<(262144 + 255) / 256, 256, 0, stream>>>(tok_w, wtok, 262144);
  cvt_bf16_kernel<<<(262144 + 255) / 256, 256, 0, stream>>>(out_w, wout, 262144);
  cvt_bf16_kernel<<<(524288 + 255) / 256, 256, 0, stream>>>(freq, freqb, 524288);

  // depthwise conv -> bf16 tokens
  dwconv_kernel<<<32768, 256, 0, stream>>>(x, dw_w, dw_b, xc);

  // xl = conv @ lin_w^T + lin_b  -> xs (bf16 tok + f32 NCHW), z (f32 tok)
  gemm_wmma<0><<<dim3(128, 8), 256, 0, stream>>>(xc, wlin, lin_b, xs_n, zbuf, xs_b, nullptr);

  // vel = xs @ vel_w^T + vel_b -> f32 NCHW
  gemm_wmma<1><<<dim3(128, 4), 256, 0, stream>>>(xs_b, wvel, vel_b, xv_n, nullptr, nullptr, nullptr);

  // token path: relu(freq @ tok_w^T + tok_b) -> cosT/sinT (C,HW)
  gemm_wmma<2><<<dim3(8, 4), 256, 0, stream>>>(freqb, wtok, tok_b, cosT, sinT, nullptr, cptr);

  // spectral pipeline -> xf token-major f32
  dct_combine_kernel<<<2048, 128, 0, stream>>>(xs_n, xv_n, cosT, sinT, aptr, xf);

  // layernorm + silu gate -> bf16 tokens
  ln_gate_kernel<<<2048, 256, 0, stream>>>(xf, zbuf, gamma, beta, gated);

  // out = gated @ out_w^T + out_b -> NCHW f32
  gemm_wmma<3><<<dim3(128, 4), 256, 0, stream>>>(gated, wout, out_b, (float*)d_out,
                                                 nullptr, nullptr, nullptr);
}
